// MambaMixer_43782896616019
// MI455X (gfx1250) — compile-verified, matrix operand order checked by reading, not run on hardware
//
#include <hip/hip_runtime.h>

// ---------------------------------------------------------------------------
// Types for CDNA5 WMMA (wave32, 16x16x32 bf16 -> f32)
// ---------------------------------------------------------------------------
typedef __bf16 bf16;
typedef __attribute__((ext_vector_type(16))) __bf16 v16bf;
typedef __attribute__((ext_vector_type(8)))  __bf16 v8bf;
typedef __attribute__((ext_vector_type(8)))  float  v8f;

union V16U { v16bf v; v8bf h[2]; };

// ---------------------------------------------------------------------------
// Problem constants (from reference)
// ---------------------------------------------------------------------------
constexpr int kH    = 32;
constexpr int kP    = 128;
constexpr int kN    = 128;
constexpr int kG    = 8;
constexpr int kK    = 4;
constexpr int kCH   = 256;
constexpr int kE    = 2048;
constexpr int kI    = kH * kP;            // 4096
constexpr int kCD   = kI + 2 * kG * kN;   // 6144
constexpr int kPROJ = kI + kCD + kH;      // 10272
constexpr int kB    = 2;
constexpr int kS    = 2048;
constexpr int kT    = kB * kS;            // 4096 tokens
constexpr int kNC   = kS / kCH;           // 8 chunks per sequence
constexpr int kGN   = kG * kN;            // 1024

// ---------------------------------------------------------------------------
// CDNA5 async global->LDS copy (ASYNCcnt-tracked, no VGPR staging)
// ---------------------------------------------------------------------------
__device__ inline void async_ld_b128(unsigned lds_off, const void* gptr) {
  asm volatile("global_load_async_to_lds_b128 %0, %1, off"
               :
               : "v"(lds_off), "v"((unsigned long long)(uintptr_t)gptr)
               : "memory");
}

__device__ inline void wait_async() {
  asm volatile("s_wait_asynccnt 0x0" ::: "memory");
}

__device__ inline unsigned lds_off_of(const void* p) {
  // generic pointers to LDS carry the LDS byte offset in the low 32 bits
  return (unsigned)(uintptr_t)p;
}

// ---------------------------------------------------------------------------
// Helpers
// ---------------------------------------------------------------------------
__device__ inline v8f vzero8() {
  v8f z;
#pragma unroll
  for (int i = 0; i < 8; ++i) z[i] = 0.f;
  return z;
}

// Load a 16x32 bf16 WMMA operand fragment from a row-major tile.
// lane 0-15: row = lane,  K runs {k0..k0+7, k0+16..k0+23}
// lane16-31: row = lane-16, K runs {k0+8..k0+15, k0+24..k0+31}
__device__ inline v16bf frag_ld(const bf16* __restrict__ base, int lane, int row_stride) {
  const int row  = lane & 15;
  const int koff = (lane & 16) ? 8 : 0;
  const bf16* p = base + (size_t)row * row_stride + koff;
  V16U u;
  u.h[0] = *(const v8bf*)(p);
  u.h[1] = *(const v8bf*)(p + 16);
  return u.v;
}

__device__ inline v8f wmma_bf16(v16bf a, v16bf b, v8f c) {
  return __builtin_amdgcn_wmma_f32_16x16x32_bf16(false, a, false, b, (short)0, c,
                                                 false, false);
}

__device__ inline float siluf(float x) { return x / (1.f + __expf(-x)); }

// ---------------------------------------------------------------------------
// fp32 -> bf16 conversion (bandwidth noise vs. the GEMMs)
// ---------------------------------------------------------------------------
__global__ __launch_bounds__(256) void cvt_f32_bf16(const float* __restrict__ in,
                                                    bf16* __restrict__ out, size_t n) {
  size_t i = (size_t)blockIdx.x * blockDim.x + threadIdx.x;
  size_t stride = (size_t)gridDim.x * blockDim.x;
  for (; i < n; i += stride) out[i] = (bf16)in[i];
}

// ---------------------------------------------------------------------------
// Generic NT GEMM:  C[M,N] = A[M,K] * B[N,K]^T   (bf16 in, fp32 out)
// Block tile 128x128, 8 waves as 2(M)x4(N), each wave 64x32 = 4x2 WMMA tiles.
// Double-buffered LDS; K tiles staged via async global->LDS DMA so the 8
// WMMAs per K-step overlap the loads; one barrier + s_wait_asynccnt per step.
// B rows beyond Nn are clamped: their products land only in accumulator
// columns that are never stored.
// ---------------------------------------------------------------------------
__global__ __launch_bounds__(256) void gemm_nt(const bf16* __restrict__ A,
                                               const bf16* __restrict__ B,
                                               float* __restrict__ C,
                                               int M, int Nn, int Kk) {
  __shared__ __align__(16) bf16 As[2][128 * 32];
  __shared__ __align__(16) bf16 Bs[2][128 * 32];

  const int tid  = threadIdx.x;
  const int lane = tid & 31;
  const int w    = tid >> 5;
  const int wm   = w & 1;   // 2 waves along M (64 rows each)
  const int wn   = w >> 1;  // 4 waves along N (32 cols each)
  const int mBlk = blockIdx.y * 128;
  const int nBlk = blockIdx.x * 128;
  const int rowL = tid >> 1;        // 0..127
  const int colL = (tid & 1) * 16;  // 0 or 16

  const int nClamp = (nBlk + rowL < Nn) ? (nBlk + rowL) : (Nn - 1);
  const bf16* srcA = A + (size_t)(mBlk + rowL) * Kk + colL;
  const bf16* srcB = B + (size_t)nClamp * Kk + colL;
  const unsigned dA = lds_off_of(&As[0][rowL * 32 + colL]);
  const unsigned dB = lds_off_of(&Bs[0][rowL * 32 + colL]);
  const unsigned bufStride = (unsigned)(128 * 32 * sizeof(bf16));

  v8f acc[4][2];
#pragma unroll
  for (int mt = 0; mt < 4; ++mt)
#pragma unroll
    for (int nt = 0; nt < 2; ++nt) acc[mt][nt] = vzero8();

  // prologue: stage K tile 0 into buffer 0
  async_ld_b128(dA,      srcA);
  async_ld_b128(dA + 16, srcA + 8);
  async_ld_b128(dB,      srcB);
  async_ld_b128(dB + 16, srcB + 8);
  wait_async();
  __syncthreads();

  int buf = 0;
  for (int k0 = 0; k0 < Kk; k0 += 32) {
    // stage next K tile into the other buffer (overlaps the WMMAs below)
    if (k0 + 32 < Kk) {
      const unsigned o = (buf ^ 1) * bufStride;
      async_ld_b128(dA + o,      srcA + k0 + 32);
      async_ld_b128(dA + o + 16, srcA + k0 + 40);
      async_ld_b128(dB + o,      srcB + k0 + 32);
      async_ld_b128(dB + o + 16, srcB + k0 + 40);
    }

    const bf16* as = As[buf];
    const bf16* bs = Bs[buf];
    v16bf af[4], bfv[2];
#pragma unroll
    for (int mt = 0; mt < 4; ++mt)
      af[mt] = frag_ld(as + (wm * 64 + mt * 16) * 32, lane, 32);
#pragma unroll
    for (int nt = 0; nt < 2; ++nt)
      bfv[nt] = frag_ld(bs + (wn * 32 + nt * 16) * 32, lane, 32);
#pragma unroll
    for (int mt = 0; mt < 4; ++mt)
#pragma unroll
      for (int nt = 0; nt < 2; ++nt)
        acc[mt][nt] = wmma_bf16(af[mt], bfv[nt], acc[mt][nt]);

    wait_async();
    __syncthreads();
    buf ^= 1;
  }

  // Store D: VGPR v, lane l -> row = base + v + 8*(l>=16), col = base + (l&15)
#pragma unroll
  for (int mt = 0; mt < 4; ++mt)
#pragma unroll
    for (int nt = 0; nt < 2; ++nt) {
      const int col = nBlk + wn * 32 + nt * 16 + (lane & 15);
      if (col >= Nn) continue;
      const int rb = mBlk + wm * 64 + mt * 16 + ((lane & 16) ? 8 : 0);
#pragma unroll
      for (int v = 0; v < 8; ++v)
        C[(size_t)(rb + v) * Nn + col] = acc[mt][nt][v];
    }
}

// ---------------------------------------------------------------------------
// dt = softplus(proj_dt + dt_bias);  abt = -exp(A_log)*dt
// ---------------------------------------------------------------------------
__global__ __launch_bounds__(256) void dtk(const float* __restrict__ proj,
                                           const float* __restrict__ dt_bias,
                                           const float* __restrict__ A_log,
                                           float* __restrict__ dt,
                                           float* __restrict__ abt) {
  int idx = blockIdx.x * 256 + threadIdx.x;
  if (idx >= kT * kH) return;
  int t = idx / kH, h = idx % kH;
  float x = proj[(size_t)t * kPROJ + kI + kCD + h] + dt_bias[h];
  float d = (x > 20.f) ? x : log1pf(__expf(x));
  dt[idx]  = d;
  abt[idx] = -__expf(A_log[h]) * d;
}

// ---------------------------------------------------------------------------
// Depthwise causal conv (K=4) + bias + SiLU, split into hs(fp32), X*dt, B, C (bf16)
// ---------------------------------------------------------------------------
__global__ __launch_bounds__(256) void convact(const float* __restrict__ proj,
                                               const float* __restrict__ cw,
                                               const float* __restrict__ cb,
                                               const float* __restrict__ dt,
                                               float* __restrict__ hs,
                                               bf16* __restrict__ xdt,
                                               bf16* __restrict__ bbf,
                                               bf16* __restrict__ cbf) {
  const int t  = blockIdx.x;
  const int ss = t & (kS - 1);  // position within sequence
  for (int ch = threadIdx.x; ch < kCD; ch += 256) {
    float acc = cb[ch];
#pragma unroll
    for (int k = 0; k < kK; ++k) {
      int sp = ss - (kK - 1) + k;
      if (sp >= 0)
        acc += proj[(size_t)(t - (kK - 1) + k) * kPROJ + kI + ch] * cw[ch * kK + k];
    }
    float v = siluf(acc);
    if (ch < kI) {
      hs[(size_t)t * kI + ch] = v;
      int h = ch >> 7;  // / kP
      xdt[(size_t)t * kI + ch] = (bf16)(v * dt[t * kH + h]);
    } else if (ch < kI + kGN) {
      bbf[(size_t)t * kGN + (ch - kI)] = (bf16)v;
    } else {
      cbf[(size_t)t * kGN + (ch - kI - kGN)] = (bf16)v;
    }
  }
}

// ---------------------------------------------------------------------------
// Per-(b,chunk,head) inclusive cumsum of A*dt over the chunk
// ---------------------------------------------------------------------------
__global__ __launch_bounds__(256) void chunkcum(const float* __restrict__ abt,
                                                float* __restrict__ acum,
                                                float* __restrict__ alast) {
  const int blk = blockIdx.x;  // (b*kNC + c)*kH + h
  const int h  = blk % kH;
  const int bc = blk / kH;
  const int c  = bc % kNC;
  const int b  = bc / kNC;
  const int t0 = b * kS + c * kCH;
  __shared__ float buf[kCH];
  const int l = threadIdx.x;
  buf[l] = abt[(size_t)(t0 + l) * kH + h];
  __syncthreads();
  for (int off = 1; off < kCH; off <<= 1) {
    float add = (l >= off) ? buf[l - off] : 0.f;
    __syncthreads();
    buf[l] += add;
    __syncthreads();
  }
  acum[(size_t)blk * kCH + l] = buf[l];
  if (l == kCH - 1) alast[blk] = buf[l];
}

// ---------------------------------------------------------------------------
// SSD intra-chunk: Y_diag = (exp(segsum) .* (C B^T)) @ X   and
// chunk states[n][p] = sum_l B[l][n]*exp(alast-acum[l])*X[l][p]
// One workgroup (8 waves) per (b,chunk,head). All GEMMs via bf16 WMMA.
// LDS arena: X^T (64KB) + acum (1KB) + max(C band+S band, Bdec^T) (64KB) ~ 129KB
// ---------------------------------------------------------------------------
__global__ __launch_bounds__(256) void ssd_diag(const bf16* __restrict__ xdt,
                                                const bf16* __restrict__ bbf,
                                                const bf16* __restrict__ cbf,
                                                const float* __restrict__ acum,
                                                float* __restrict__ yacc,
                                                float* __restrict__ states) {
  __shared__ __align__(16) char arena[132096];
  bf16*  xst = (bf16*)(arena);            // [128][256] X^T, p-major
  float* acs = (float*)(arena + 65536);   // [256]
  bf16*  cs  = (bf16*)(arena + 66560);    // [64][128]  C band
  bf16*  st  = (bf16*)(arena + 82944);    // [64][256]  masked scores (bf16)
  bf16*  bt  = (bf16*)(arena + 66560);    // [128][256] (B*decay)^T, reuses cs+st

  const int tid  = threadIdx.x;
  const int lane = tid & 31;
  const int w    = tid >> 5;
  const int blk  = blockIdx.x;  // (b*kNC + c)*kH + h
  const int h  = blk % kH;
  const int bc = blk / kH;
  const int c  = bc % kNC;
  const int b  = bc / kNC;
  const int g  = h % kG;
  const int t0 = b * kS + c * kCH;

  acs[tid] = acum[(size_t)blk * kCH + tid];

  // Stage X^T (thread l streams its row, scatters into p-major LDS)
  {
    const bf16* src = xdt + (size_t)(t0 + tid) * kI + h * kP;
#pragma unroll
    for (int p0 = 0; p0 < kP; p0 += 8) {
      v8bf v = *(const v8bf*)(src + p0);
#pragma unroll
      for (int j = 0; j < 8; ++j) xst[(size_t)(p0 + j) * kCH + tid] = v[j];
    }
  }
  __syncthreads();
  const float a_last = acs[kCH - 1];

  const int wm = w & 1;   // 2 waves along rows
  const int wn = w >> 1;  // 4 waves along cols

  // ---- 4 row bands of 64 for the 256x256 score matrix ----
  for (int r = 0; r < 4; ++r) {
    {  // stage C band [64][128] via async global->LDS DMA (contiguous copy)
      const int row = tid >> 2;
      const int seg = (tid & 3) * 32;
      const bf16* src = cbf + (size_t)(t0 + r * 64 + row) * kGN + g * kN + seg;
      const unsigned dst = lds_off_of(cs + row * kN + seg);
      async_ld_b128(dst,      src);
      async_ld_b128(dst + 16, src + 8);
      async_ld_b128(dst + 32, src + 16);
      async_ld_b128(dst + 48, src + 24);
      wait_async();
    }
    __syncthreads();

    // S_band = C_band(64x128) @ B^T(128x256); B fragments streamed from global
    v8f accS[2][4];
#pragma unroll
    for (int mt = 0; mt < 2; ++mt)
#pragma unroll
      for (int nt = 0; nt < 4; ++nt) accS[mt][nt] = vzero8();

#pragma unroll
    for (int k0 = 0; k0 < kN; k0 += 32) {
      v16bf af[2];
      af[0] = frag_ld(cs + (wm * 32 + 0) * kN + k0, lane, kN);
      af[1] = frag_ld(cs + (wm * 32 + 16) * kN + k0, lane, kN);
      v16bf bfr[4];
#pragma unroll
      for (int nt = 0; nt < 4; ++nt) {
        const int lcol = wn * 64 + nt * 16 + (lane & 15);
        const int kk = k0 + ((lane & 16) ? 8 : 0);
        const bf16* p = bbf + (size_t)(t0 + lcol) * kGN + g * kN + kk;
        V16U u;
        u.h[0] = *(const v8bf*)(p);
        u.h[1] = *(const v8bf*)(p + 16);
        bfr[nt] = u.v;
      }
#pragma unroll
      for (int mt = 0; mt < 2; ++mt)
#pragma unroll
        for (int nt = 0; nt < 4; ++nt)
          accS[mt][nt] = wmma_bf16(af[mt], bfr[nt], accS[mt][nt]);
    }

    // mask (lower triangular) + segment decay, restage as bf16
#pragma unroll
    for (int mt = 0; mt < 2; ++mt)
#pragma unroll
      for (int nt = 0; nt < 4; ++nt) {
        const int lc = wn * 64 + nt * 16 + (lane & 15);
        const int rbase = wm * 32 + mt * 16 + ((lane & 16) ? 8 : 0);
#pragma unroll
        for (int v = 0; v < 8; ++v) {
          const int lrl = rbase + v;
          const int lr  = r * 64 + lrl;
          float val = (lc <= lr) ? accS[mt][nt][v] * __expf(acs[lr] - acs[lc]) : 0.f;
          st[(size_t)lrl * kCH + lc] = (bf16)val;
        }
      }
    __syncthreads();

    // Y_band = S_band(64x256) @ X(256x128)
    v8f accY[2][2];
#pragma unroll
    for (int mt = 0; mt < 2; ++mt)
#pragma unroll
      for (int nt = 0; nt < 2; ++nt) accY[mt][nt] = vzero8();

#pragma unroll
    for (int k0 = 0; k0 < kCH; k0 += 32) {
      v16bf af[2];
      af[0] = frag_ld(st + (wm * 32 + 0) * kCH + k0, lane, kCH);
      af[1] = frag_ld(st + (wm * 32 + 16) * kCH + k0, lane, kCH);
      v16bf bfr[2];
#pragma unroll
      for (int nt = 0; nt < 2; ++nt)
        bfr[nt] = frag_ld(xst + (wn * 32 + nt * 16) * kCH + k0, lane, kCH);
#pragma unroll
      for (int mt = 0; mt < 2; ++mt)
#pragma unroll
        for (int nt = 0; nt < 2; ++nt)
          accY[mt][nt] = wmma_bf16(af[mt], bfr[nt], accY[mt][nt]);
    }

#pragma unroll
    for (int mt = 0; mt < 2; ++mt)
#pragma unroll
      for (int nt = 0; nt < 2; ++nt) {
        const int p = wn * 32 + nt * 16 + (lane & 15);
        const int rb = r * 64 + wm * 32 + mt * 16 + ((lane & 16) ? 8 : 0);
#pragma unroll
        for (int v = 0; v < 8; ++v)
          yacc[(size_t)(t0 + rb + v) * kI + h * kP + p] = accY[mt][nt][v];
      }
    __syncthreads();  // before reusing cs/st in next band
  }

  // ---- chunk states: bt[n][l] = B[l][n]*exp(alast-acum[l]) ----
  {
    const float dec = __expf(a_last - acs[tid]);
    const bf16* src = bbf + (size_t)(t0 + tid) * kGN + g * kN;
#pragma unroll
    for (int n0 = 0; n0 < kN; n0 += 8) {
      v8bf v = *(const v8bf*)(src + n0);
#pragma unroll
      for (int j = 0; j < 8; ++j)
        bt[(size_t)(n0 + j) * kCH + tid] = (bf16)((float)v[j] * dec);
    }
  }
  __syncthreads();

  // states[n][p] = bt(128x256) @ X(256x128); each wave owns 16 n-rows
  {
    v8f acc[8];
#pragma unroll
    for (int nt = 0; nt < 8; ++nt) acc[nt] = vzero8();
#pragma unroll
    for (int k0 = 0; k0 < kCH; k0 += 32) {
      v16bf af = frag_ld(bt + (w * 16) * kCH + k0, lane, kCH);
#pragma unroll
      for (int nt = 0; nt < 8; ++nt) {
        v16bf bfr = frag_ld(xst + (nt * 16) * kCH + k0, lane, kCH);
        acc[nt] = wmma_bf16(af, bfr, acc[nt]);
      }
    }
    const size_t base = (size_t)blk * kN * kP;
#pragma unroll
    for (int nt = 0; nt < 8; ++nt) {
      const int p = nt * 16 + (lane & 15);
      const int nb = w * 16 + ((lane & 16) ? 8 : 0);
#pragma unroll
      for (int v = 0; v < 8; ++v)
        states[base + (size_t)(nb + v) * kP + p] = acc[nt][v];
    }
  }
}

// ---------------------------------------------------------------------------
// Inter-chunk sequential scan: prev_{c+1} = exp(alast_c)*prev_c + states_c
// states_prev[c] = prev_c. One block per (b,h); 8 steps; state in registers.
// ---------------------------------------------------------------------------
__global__ __launch_bounds__(256) void statescan(const float* __restrict__ states,
                                                 const float* __restrict__ alast,
                                                 float* __restrict__ sprev) {
  const int blk = blockIdx.x;  // b*kH + h
  const int h = blk % kH;
  const int b = blk / kH;
  float pr[64];
#pragma unroll
  for (int j = 0; j < 64; ++j) pr[j] = 0.f;
  for (int c = 0; c < kNC; ++c) {
    const int bch = (b * kNC + c) * kH + h;
    const size_t base = (size_t)bch * kN * kP;
    const float ef = __expf(alast[bch]);
    for (int j = 0; j < 64; ++j) {
      const size_t e = base + threadIdx.x + 256 * j;
      sprev[e] = pr[j];
      pr[j] = ef * pr[j] + states[e];
    }
  }
}

// ---------------------------------------------------------------------------
// SSD inter-chunk contribution + D residual:
// y += (C .* exp(acum)) @ prev_state^T + D[h]*hs
// ---------------------------------------------------------------------------
__global__ __launch_bounds__(256) void ssd_off(const bf16* __restrict__ cbf,
                                               const float* __restrict__ sprev,
                                               const float* __restrict__ acum,
                                               const float* __restrict__ hs,
                                               const float* __restrict__ Dp,
                                               float* __restrict__ yio) {
  __shared__ __align__(16) char arena[99328];
  bf16* spT = (bf16*)(arena);            // [P][N] prev-state transposed
  bf16* cs2 = (bf16*)(arena + 32768);    // [CH][N] C * state_decay

  const int tid  = threadIdx.x;
  const int lane = tid & 31;
  const int w    = tid >> 5;
  const int blk  = blockIdx.x;  // (b*kNC + c)*kH + h
  const int h  = blk % kH;
  const int bc = blk / kH;
  const int c  = bc % kNC;
  const int b  = bc / kNC;
  const int g  = h % kG;
  const int t0 = b * kS + c * kCH;

  // prev state, transposed to [p][n] so B-fragments read contiguous n
  {
    const size_t sbase = (size_t)blk * kN * kP;
    for (int idx = tid; idx < kN * kP; idx += 256) {
      const int p = idx & (kP - 1);
      const int n = idx >> 7;
      spT[(size_t)p * kN + n] = (bf16)sprev[sbase + (size_t)n * kP + p];
    }
  }
  // C scaled by exp(acum[l])
  {
    const float sd = __expf(acum[(size_t)blk * kCH + tid]);
    const bf16* src = cbf + (size_t)(t0 + tid) * kGN + g * kN;
#pragma unroll
    for (int n0 = 0; n0 < kN; n0 += 8) {
      v8bf v = *(const v8bf*)(src + n0);
      v8bf o;
#pragma unroll
      for (int j = 0; j < 8; ++j) o[j] = (bf16)((float)v[j] * sd);
      *(v8bf*)(cs2 + (size_t)tid * kN + n0) = o;
    }
  }
  __syncthreads();

  const float Dh = Dp[h];
  // wave w owns rows [w*32, w*32+32); two passes over 128 p-columns
  for (int pass = 0; pass < 2; ++pass) {
    v8f acc[2][4];
#pragma unroll
    for (int mt = 0; mt < 2; ++mt)
#pragma unroll
      for (int nt = 0; nt < 4; ++nt) acc[mt][nt] = vzero8();

#pragma unroll
    for (int k0 = 0; k0 < kN; k0 += 32) {
      v16bf af[2];
      af[0] = frag_ld(cs2 + (w * 32 + 0) * kN + k0, lane, kN);
      af[1] = frag_ld(cs2 + (w * 32 + 16) * kN + k0, lane, kN);
      v16bf bfr[4];
#pragma unroll
      for (int nt = 0; nt < 4; ++nt)
        bfr[nt] = frag_ld(spT + (pass * 64 + nt * 16) * kN + k0, lane, kN);
#pragma unroll
      for (int mt = 0; mt < 2; ++mt)
#pragma unroll
        for (int nt = 0; nt < 4; ++nt)
          acc[mt][nt] = wmma_bf16(af[mt], bfr[nt], acc[mt][nt]);
    }

#pragma unroll
    for (int mt = 0; mt < 2; ++mt)
#pragma unroll
      for (int nt = 0; nt < 4; ++nt) {
        const int p = pass * 64 + nt * 16 + (lane & 15);
        const int rb = w * 32 + mt * 16 + ((lane & 16) ? 8 : 0);
#pragma unroll
        for (int v = 0; v < 8; ++v) {
          const size_t idx = (size_t)(t0 + rb + v) * kI + h * kP + p;
          yio[idx] = yio[idx] + acc[mt][nt][v] + Dh * hs[idx];
        }
      }
  }
}

// ---------------------------------------------------------------------------
// Gated group RMSNorm: h = y*silu(gate); per-512 group rms; *norm_w -> bf16
// ---------------------------------------------------------------------------
__global__ __launch_bounds__(256) void gatenorm(const float* __restrict__ y,
                                                const float* __restrict__ proj,
                                                const float* __restrict__ nw,
                                                bf16* __restrict__ ybf) {
  const int t   = blockIdx.x >> 3;
  const int grp = blockIdx.x & 7;
  const int i0  = grp * 512;
  const int tid = threadIdx.x;
  __shared__ float red[256];

  const size_t base = (size_t)t * kI + i0;
  const size_t pbase = (size_t)t * kPROJ + i0;
  float y0 = y[base + tid],        y1 = y[base + tid + 256];
  float g0 = proj[pbase + tid],    g1 = proj[pbase + tid + 256];
  float h0 = y0 * siluf(g0);
  float h1 = y1 * siluf(g1);
  red[tid] = h0 * h0 + h1 * h1;
  __syncthreads();
  for (int s = 128; s > 0; s >>= 1) {
    if (tid < s) red[tid] += red[tid + s];
    __syncthreads();
  }
  const float r = rsqrtf(red[0] / 512.f + 1e-6f);
  ybf[base + tid]       = (bf16)(h0 * r * nw[i0 + tid]);
  ybf[base + tid + 256] = (bf16)(h1 * r * nw[i0 + tid + 256]);
}

// ---------------------------------------------------------------------------
// Host-side orchestration
// ---------------------------------------------------------------------------
extern "C" void kernel_launch(void* const* d_in, const int* in_sizes, int n_in,
                              void* d_out, int out_size, void* d_ws, size_t ws_size,
                              hipStream_t stream) {
  (void)in_sizes; (void)n_in; (void)out_size; (void)ws_size;
  const float* x    = (const float*)d_in[0];
  const float* w1   = (const float*)d_in[1];
  const float* cw   = (const float*)d_in[2];
  const float* cb   = (const float*)d_in[3];
  const float* dtb  = (const float*)d_in[4];
  const float* alog = (const float*)d_in[5];
  const float* Dp   = (const float*)d_in[6];
  const float* nw   = (const float*)d_in[7];
  const float* w2   = (const float*)d_in[8];
  float* out = (float*)d_out;

  char* p = (char*)d_ws;
  auto alloc = [&](size_t bytes) -> char* {
    char* r = p;
    p += (bytes + 255) & ~(size_t)255;
    return r;
  };
  bf16*  xbf    = (bf16*)alloc((size_t)kT * kE * 2);
  bf16*  w1bf   = (bf16*)alloc((size_t)kPROJ * kE * 2);
  bf16*  w2bf   = (bf16*)alloc((size_t)kE * kI * 2);
  float* proj   = (float*)alloc((size_t)kT * kPROJ * 4);
  float* dtA    = (float*)alloc((size_t)kT * kH * 4);
  float* abt    = (float*)alloc((size_t)kT * kH * 4);
  float* acum   = (float*)alloc((size_t)kB * kNC * kH * kCH * 4);
  float* alast  = (float*)alloc((size_t)kB * kNC * kH * 4);
  float* hs     = (float*)alloc((size_t)kT * kI * 4);
  bf16*  xdt    = (bf16*)alloc((size_t)kT * kI * 2);
  bf16*  bbf    = (bf16*)alloc((size_t)kT * kGN * 2);
  bf16*  cbf    = (bf16*)alloc((size_t)kT * kGN * 2);
  float* states = (float*)alloc((size_t)kB * kNC * kH * kN * kP * 4);
  float* sprev  = (float*)alloc((size_t)kB * kNC * kH * kN * kP * 4);
  float* yacc   = (float*)alloc((size_t)kT * kI * 4);
  bf16*  ybf    = (bf16*)alloc((size_t)kT * kI * 2);

  // 1) fp32 -> bf16 conversions
  cvt_f32_bf16<<<4096, 256, 0, stream>>>(x,  xbf,  (size_t)kT * kE);
  cvt_f32_bf16<<<8192, 256, 0, stream>>>(w1, w1bf, (size_t)kPROJ * kE);
  cvt_f32_bf16<<<4096, 256, 0, stream>>>(w2, w2bf, (size_t)kE * kI);

  // 2) in_proj GEMM: proj[T, PROJ] = x @ w1^T
  dim3 g1((kPROJ + 127) / 128, kT / 128);
  gemm_nt<<<g1, 256, 0, stream>>>(xbf, w1bf, proj, kT, kPROJ, kE);

  // 3) dt / A*dt, then conv+SiLU+split
  dtk<<<(kT * kH + 255) / 256, 256, 0, stream>>>(proj, dtb, alog, dtA, abt);
  convact<<<kT, 256, 0, stream>>>(proj, cw, cb, dtA, hs, xdt, bbf, cbf);

  // 4) per-chunk cumulative decays
  chunkcum<<<kB * kNC * kH, kCH, 0, stream>>>(abt, acum, alast);

  // 5) intra-chunk SSD (Y_diag + chunk states), WMMA
  ssd_diag<<<kB * kNC * kH, 256, 0, stream>>>(xdt, bbf, cbf, acum, yacc, states);

  // 6) inter-chunk state recurrence
  statescan<<<kB * kH, 256, 0, stream>>>(states, alast, sprev);

  // 7) inter-chunk contribution + D residual, WMMA
  ssd_off<<<kB * kNC * kH, 256, 0, stream>>>(cbf, sprev, acum, hs, Dp, yacc);

  // 8) gated group RMSNorm -> bf16
  gatenorm<<<kT * 8, 256, 0, stream>>>(yacc, proj, nw, ybf);

  // 9) out_proj GEMM: out[T, E] = ybf @ w2^T
  dim3 g2(kE / 128, kT / 128);
  gemm_nt<<<g2, 256, 0, stream>>>(ybf, w2bf, out, kT, kE, kI);
}
